// MaskedSelfAttention_62869731279511
// MI455X (gfx1250) — matrix operand order repaired and checked by reading, not compiled, not run on hardware
//
#include <hip/hip_runtime.h>
#include <hip/hip_bf16.h>

// ---------------------------------------------------------------------------
// Problem constants (reference: B=2, N=2048, C=1024, H=16, D=64)
// ---------------------------------------------------------------------------
constexpr int BB = 2;
constexpr int NN = 2048;
constexpr int CC = 1024;
constexpr int HH = 16;
constexpr int DD = 64;
constexpr int BN = BB * NN;          // 4096 token rows

typedef __attribute__((ext_vector_type(16))) __bf16 v16bf;
typedef __attribute__((ext_vector_type(8)))  float  v8f;
typedef int i32x4 __attribute__((vector_size(16)));   // matches builtin pointee

union Frag { v16bf v; uint4 u[2]; };

__device__ __forceinline__ unsigned short f2bf(float f) {
  unsigned u = __float_as_uint(f);
  u += 0x7fffu + ((u >> 16) & 1u);   // round-to-nearest-even
  return (unsigned short)(u >> 16);
}

__device__ __forceinline__ uint4 ldg16(const unsigned short* p) {
  return *(const uint4*)p;
}

__device__ __forceinline__ v8f wmma_bf16(const Frag& a, const Frag& b, v8f c) {
  return __builtin_amdgcn_wmma_f32_16x16x32_bf16(false, a.v, false, b.v,
                                                 (short)0, c, false, false);
}

// ---------------------------------------------------------------------------
// gfx1250 async global->LDS copy (ASYNCcnt-tracked), with portable fallback
// ---------------------------------------------------------------------------
#if __has_builtin(__builtin_amdgcn_global_load_async_to_lds_b128)
#define USE_ASYNC_LDS 1
#else
#define USE_ASYNC_LDS 0
#endif

__device__ __forceinline__ void cp16_g2s(const unsigned short* g,
                                         unsigned short* l) {
#if USE_ASYNC_LDS
  __builtin_amdgcn_global_load_async_to_lds_b128(
      (__attribute__((address_space(1))) i32x4*)g,
      (__attribute__((address_space(3))) i32x4*)l, 0, 0);
#else
  *(uint4*)l = *(const uint4*)g;
#endif
}

__device__ __forceinline__ void wait_async_le8() {
#if USE_ASYNC_LDS
#if __has_builtin(__builtin_amdgcn_s_wait_asynccnt)
  __builtin_amdgcn_s_wait_asynccnt(8);
#else
  asm volatile("s_wait_asynccnt 8" ::: "memory");
#endif
#endif
}

__device__ __forceinline__ void wait_async_le0() {
#if USE_ASYNC_LDS
#if __has_builtin(__builtin_amdgcn_s_wait_asynccnt)
  __builtin_amdgcn_s_wait_asynccnt(0);
#else
  asm volatile("s_wait_asynccnt 0" ::: "memory");
#endif
#endif
}

// ---------------------------------------------------------------------------
// fp32 -> bf16 conversion (grid-stride)
// ---------------------------------------------------------------------------
__global__ void k_f32_to_bf16(const float* __restrict__ in,
                              unsigned short* __restrict__ out, int n) {
  int i = blockIdx.x * blockDim.x + threadIdx.x;
  int stride = gridDim.x * blockDim.x;
  for (; i < n; i += stride) out[i] = f2bf(in[i]);
}

// ---------------------------------------------------------------------------
// LDS-staged, double-buffered WMMA GEMM: block tile 64x64, 4 waves, K-step 64.
//   out = A[M x K](bf16) @ W[Ncols x K](bf16)^T + bias
// MODE 0: plain fp32 epilogue (projection GEMM)
// MODE 1: fused QKV epilogue — bias, q/k RMS-norm (+ D^-0.5 on q) in fp32,
//         bf16 stores to qbuf/kbuf (B,H,N,D) and v transposed to vt (B,H,D,N).
// ---------------------------------------------------------------------------
constexpr int LDSP = 72;             // padded LDS row stride (bank-spread, 16B mult)

template <int MODE>
__global__ __launch_bounds__(128)
void k_gemm(const unsigned short* __restrict__ A,
            const unsigned short* __restrict__ W,
            const float* __restrict__ bias,
            float* __restrict__ outf,
            const float* __restrict__ qn_w, const float* __restrict__ kn_w,
            unsigned short* __restrict__ qbuf, unsigned short* __restrict__ kbuf,
            unsigned short* __restrict__ vt,
            int Ncols, int K) {
  __shared__ __align__(16) unsigned short Asm[2][64 * LDSP];
  __shared__ __align__(16) unsigned short Bsm[2][64 * LDSP];

  const int tid  = threadIdx.x;
  const int wave = tid >> 5;
  const int lane = tid & 31;
  const int hf   = lane >> 4;
  const int nloc = lane & 15;
  const int mbase = blockIdx.x * 64;
  const int nbase = blockIdx.y * 64;

  // 512 16B-chunks per 64x64 tile; 4 per thread per tile
  auto fill = [&](int buf, int kk) {
#pragma unroll
    for (int i = 0; i < 4; ++i) {
      const int c = tid + i * 128;
      const int row = c >> 3, col8 = c & 7;
      cp16_g2s(A + (size_t)(mbase + row) * K + kk + col8 * 8,
               &Asm[buf][row * LDSP + col8 * 8]);
    }
#pragma unroll
    for (int i = 0; i < 4; ++i) {
      const int c = tid + i * 128;
      const int row = c >> 3, col8 = c & 7;
      cp16_g2s(W + (size_t)(nbase + row) * K + kk + col8 * 8,
               &Bsm[buf][row * LDSP + col8 * 8]);
    }
  };

  v8f acc[4] = {};
  fill(0, 0);                               // prime stage 0 (8 async ops/thread)

  for (int kk = 0; kk < K; kk += 64) {
    const int buf = (kk >> 6) & 1;
    const bool more = (kk + 64) < K;
    if (more) fill(buf ^ 1, kk + 64);       // prefetch next stage
    if (more) wait_async_le8();             // in-order: current stage landed
    else      wait_async_le0();
    __syncthreads();

    const unsigned short* As = Asm[buf];
    const unsigned short* Bs = Bsm[buf];
#pragma unroll
    for (int ks = 0; ks < 64; ks += 32) {
      Frag a;
      const unsigned short* ar = As + (wave * 16 + nloc) * LDSP + ks;
      a.u[0] = *(const uint4*)(ar + hf * 16);
      a.u[1] = *(const uint4*)(ar + hf * 16 + 8);
#pragma unroll
      for (int t = 0; t < 4; ++t) {
        const unsigned short* br = Bs + (t * 16 + nloc) * LDSP + ks + hf * 16;
        Frag b;
        b.u[0] = *(const uint4*)br;
        b.u[1] = *(const uint4*)(br + 8);
        acc[t] = wmma_bf16(a, b, acc[t]);
      }
    }
    __syncthreads();                        // tile consumed; safe to refill
  }

  // bias add (fp32)
#pragma unroll
  for (int t = 0; t < 4; ++t) {
    const float bv = bias[nbase + t * 16 + nloc];
#pragma unroll
    for (int r = 0; r < 8; ++r) acc[t][r] += bv;
  }

  if (MODE == 0) {
#pragma unroll
    for (int t = 0; t < 4; ++t) {
      const int col = nbase + t * 16 + nloc;
#pragma unroll
      for (int r = 0; r < 8; ++r) {
        const int row = mbase + wave * 16 + r + 8 * hf;
        outf[(size_t)row * Ncols + col] = acc[t][r];
      }
    }
    return;
  }

  // ---- MODE 1: fused q/k RMS-norm + v transpose ----
  const int sidx = nbase / CC;              // 0=q, 1=k, 2=v
  const int h    = (nbase % CC) / DD;
  const int b    = mbase / NN;              // 64 | NN so constant per block
  const int n0   = mbase % NN;
  const int bh   = b * HH + h;

  if (sidx < 2) {
    const float* nw = (sidx == 0) ? qn_w : kn_w;
    unsigned short* dst = (sidx == 0) ? qbuf : kbuf;
    const float post = (sidx == 0) ? 0.125f : 1.0f;   // D^-0.5 folded into q
#pragma unroll
    for (int r = 0; r < 8; ++r) {
      float ss = 0.0f;
#pragma unroll
      for (int t = 0; t < 4; ++t) ss += acc[t][r] * acc[t][r];
#pragma unroll
      for (int off = 1; off < 16; off <<= 1) ss += __shfl_xor(ss, off, 32);
      const float rr = rsqrtf(ss * (1.0f / DD) + 1e-6f) * post;
      const int tok = wave * 16 + r + 8 * hf;
      unsigned short* row = dst + ((size_t)bh * NN + n0 + tok) * DD;
#pragma unroll
      for (int t = 0; t < 4; ++t)
        row[t * 16 + nloc] = f2bf(acc[t][r] * rr * nw[t * 16 + nloc]);
    }
  } else {
    // transpose 64(tok) x 64(d) tile through LDS -> coalesced (B,H,D,N) stores
    unsigned short* T = &Asm[0][0];         // 64 x LDSP, free after main loop
#pragma unroll
    for (int r = 0; r < 8; ++r) {
      const int tok = wave * 16 + r + 8 * hf;
#pragma unroll
      for (int t = 0; t < 4; ++t)
        T[(t * 16 + nloc) * LDSP + tok] = f2bf(acc[t][r]);
    }
    __syncthreads();
    const int d = tid >> 1, seg = tid & 1;  // 2 threads per d-row, 32 tokens each
    const unsigned short* srcr = T + d * LDSP + seg * 32;
    unsigned short* dstr = vt + ((size_t)bh * DD + d) * NN + n0 + seg * 32;
#pragma unroll
    for (int j = 0; j < 4; ++j)
      *(uint4*)(dstr + j * 8) = *(const uint4*)(srcr + j * 8);
  }
}

// ---------------------------------------------------------------------------
// Flash attention: one wave per (b,h,16-query tile), streaming 32-key blocks.
// ---------------------------------------------------------------------------
__global__ __launch_bounds__(32)
void k_attention(const unsigned short* __restrict__ qb,
                 const unsigned short* __restrict__ kb,
                 const unsigned short* __restrict__ vt,
                 const int* __restrict__ mask,
                 unsigned short* __restrict__ aout) {
  __shared__ __align__(16) unsigned short Plds[16 * 32];

  const int qtile = blockIdx.x, h = blockIdx.y, b = blockIdx.z;
  const int bh = b * HH + h;
  const int lane = threadIdx.x;
  const int hf   = lane >> 4;
  const int nloc = lane & 15;
  const int qbase = qtile * 16;

  const unsigned short* qrow = qb + ((size_t)bh * NN + qbase + nloc) * DD;
  Frag a0, a1;
  a0.u[0] = ldg16(qrow + hf * 16);
  a0.u[1] = ldg16(qrow + hf * 16 + 8);
  a1.u[0] = ldg16(qrow + 32 + hf * 16);
  a1.u[1] = ldg16(qrow + 32 + hf * 16 + 8);

  float m[8], l[8];
  v8f o[4] = {};
#pragma unroll
  for (int r = 0; r < 8; ++r) { m[r] = -1e30f; l[r] = 0.0f; }

  for (int kb32 = 0; kb32 < NN; kb32 += 32) {
    const unsigned short* krow0 = kb + ((size_t)bh * NN + kb32 + nloc) * DD;
    const unsigned short* krow1 = krow0 + 16 * DD;
    Frag b00, b01, b10, b11;
    b00.u[0] = ldg16(krow0 + hf * 16);      b00.u[1] = ldg16(krow0 + hf * 16 + 8);
    b01.u[0] = ldg16(krow0 + 32 + hf * 16); b01.u[1] = ldg16(krow0 + 32 + hf * 16 + 8);
    b10.u[0] = ldg16(krow1 + hf * 16);      b10.u[1] = ldg16(krow1 + hf * 16 + 8);
    b11.u[0] = ldg16(krow1 + 32 + hf * 16); b11.u[1] = ldg16(krow1 + 32 + hf * 16 + 8);

    v8f s0 = {}, s1 = {};
    s0 = wmma_bf16(a0, b00, s0); s0 = wmma_bf16(a1, b01, s0);
    s1 = wmma_bf16(a0, b10, s1); s1 = wmma_bf16(a1, b11, s1);

    const bool keep0 = mask[b * NN + kb32 + nloc] != 0;
    const bool keep1 = mask[b * NN + kb32 + 16 + nloc] != 0;
#pragma unroll
    for (int r = 0; r < 8; ++r) {
      if (!keep0) s0[r] = -1e9f;
      if (!keep1) s1[r] = -1e9f;
    }

#pragma unroll
    for (int r = 0; r < 8; ++r) {
      float tmax = fmaxf(s0[r], s1[r]);
#pragma unroll
      for (int off = 1; off < 16; off <<= 1)
        tmax = fmaxf(tmax, __shfl_xor(tmax, off, 32));
      const float mnew  = fmaxf(m[r], tmax);
      const float alpha = __expf(m[r] - mnew);
      const float p0 = __expf(s0[r] - mnew);
      const float p1 = __expf(s1[r] - mnew);
      float rs = p0 + p1;
#pragma unroll
      for (int off = 1; off < 16; off <<= 1) rs += __shfl_xor(rs, off, 32);
      l[r] = l[r] * alpha + rs;
      m[r] = mnew;
      s0[r] = p0; s1[r] = p1;
#pragma unroll
      for (int t = 0; t < 4; ++t) o[t][r] *= alpha;
    }

    const int rowoff = 8 * hf;
#pragma unroll
    for (int r = 0; r < 8; ++r) {
      Plds[(r + rowoff) * 32 + nloc]      = f2bf(s0[r]);
      Plds[(r + rowoff) * 32 + 16 + nloc] = f2bf(s1[r]);
    }
    asm volatile("s_wait_dscnt 0" ::: "memory");
    Frag pf;
    pf.u[0] = *(const uint4*)&Plds[nloc * 32 + hf * 16];
    pf.u[1] = *(const uint4*)&Plds[nloc * 32 + hf * 16 + 8];

#pragma unroll
    for (int t = 0; t < 4; ++t) {
      const unsigned short* vrow =
          vt + ((size_t)bh * DD + t * 16 + nloc) * NN + kb32 + hf * 16;
      Frag bv;
      bv.u[0] = ldg16(vrow);
      bv.u[1] = ldg16(vrow + 8);
      o[t] = wmma_bf16(pf, bv, o[t]);
    }
  }

#pragma unroll
  for (int r = 0; r < 8; ++r) {
    const float inv = 1.0f / l[r];
    const int row = qbase + r + 8 * hf;
    unsigned short* orow = aout + ((size_t)(b * NN + row)) * CC + h * DD;
#pragma unroll
    for (int t = 0; t < 4; ++t)
      orow[t * 16 + nloc] = f2bf(o[t][r] * inv);
  }
}

// ---------------------------------------------------------------------------
// Host launcher
// ---------------------------------------------------------------------------
extern "C" void kernel_launch(void* const* d_in, const int* in_sizes, int n_in,
                              void* d_out, int out_size, void* d_ws, size_t ws_size,
                              hipStream_t stream) {
  const float* x      = (const float*)d_in[0];
  const int*   mask   = (const int*)d_in[1];
  const float* qkv_w  = (const float*)d_in[2];
  const float* qkv_b  = (const float*)d_in[3];
  const float* proj_w = (const float*)d_in[4];
  const float* proj_b = (const float*)d_in[5];
  const float* qn_w   = (const float*)d_in[6];
  const float* kn_w   = (const float*)d_in[7];
  float* out = (float*)d_out;

  // workspace carve-up (~48 MB, all bf16)
  unsigned short* xbf  = (unsigned short*)d_ws;                 // BN*C
  unsigned short* wqkv = xbf  + (size_t)BN * CC;                // 3C*C
  unsigned short* wprj = wqkv + (size_t)(3 * CC) * CC;          // C*C
  unsigned short* qbuf = wprj + (size_t)CC * CC;                // B*H*N*D
  unsigned short* kbuf = qbuf + (size_t)BN * CC;
  unsigned short* vt   = kbuf + (size_t)BN * CC;
  unsigned short* aout = vt   + (size_t)BN * CC;

  k_f32_to_bf16<<<512, 256, 0, stream>>>(x, xbf, BN * CC);
  k_f32_to_bf16<<<512, 256, 0, stream>>>(qkv_w, wqkv, 3 * CC * CC);
  k_f32_to_bf16<<<512, 256, 0, stream>>>(proj_w, wprj, CC * CC);

  // fused: qkv = x @ qkv_w^T + b, RMS-norm q/k, scale q, transpose v
  k_gemm<1><<<dim3(BN / 64, (3 * CC) / 64), 128, 0, stream>>>(
      xbf, wqkv, qkv_b, nullptr, qn_w, kn_w, qbuf, kbuf, vt, 3 * CC, CC);

  // flash attention -> aout (B,N,C) bf16
  k_attention<<<dim3(NN / 16, HH, BB), 32, 0, stream>>>(qbuf, kbuf, vt,
                                                        mask, aout);

  // out = attn_out @ proj_w^T + proj_b (fp32)
  k_gemm<0><<<dim3(BN / 64, CC / 64), 128, 0, stream>>>(
      aout, wprj, proj_b, out, nullptr, nullptr, nullptr, nullptr, nullptr,
      CC, CC);
}
